// EtaGNN_64235530879430
// MI455X (gfx1250) — compile-verified, matrix operand order checked by reading, not run on hardware
//
#include <hip/hip_runtime.h>
#include <hip/hip_bf16.h>

typedef __attribute__((ext_vector_type(16))) _Float16 v16h;
typedef __attribute__((ext_vector_type(8)))  _Float16 v8h;
typedef __attribute__((ext_vector_type(8)))  float    v8f;

#define NNODES 50000
#define NEDGES 800000
#define NQUERY 400000
#define INDIM  128
#define HIDDIM 64

// ---------------------------------------------------------------------------
// Fragment loaders (wave32, 16-bit A/B layout per CDNA5 ISA 7.12.2):
// lane L holds row (L & 15); with half = L>>4:
//   f[0..7]  = K[kbase + half*8 + 0..7]
//   f[8..15] = K[kbase + 16 + half*8 + 0..7]
// Works for both global and LDS source pointers (addrspace inferred after
// inlining; LDS path lowers to ds_load_b128).
// ---------------------------------------------------------------------------
__device__ __forceinline__ v16h frag_f16(const _Float16* __restrict__ row,
                                         int kbase, int half) {
  const v8h* q = reinterpret_cast<const v8h*>(row + kbase + half * 8);
  v8h lo = q[0];
  v8h hi = q[2];   // +16 halves
  v16h r;
#pragma unroll
  for (int i = 0; i < 8; ++i) { r[i] = lo[i]; r[8 + i] = hi[i]; }
  return r;
}

__device__ __forceinline__ v16h frag_f32(const float* __restrict__ row,
                                         int kbase, int half, float scale) {
  const float* a = row + kbase + half * 8;
  v16h r;
#pragma unroll
  for (int i = 0; i < 8; ++i) r[i]     = (_Float16)(a[i] * scale);
#pragma unroll
  for (int i = 0; i < 8; ++i) r[8 + i] = (_Float16)(a[16 + i] * scale);
  return r;
}

// Cooperative block copy of a weight matrix (global f16 -> LDS), b128 chunks.
__device__ __forceinline__ void stage_lds(_Float16* dst, const _Float16* src,
                                          int halves, int tid, int nthreads) {
  v8h* d = reinterpret_cast<v8h*>(dst);
  const v8h* s = reinterpret_cast<const v8h*>(src);
  int n = halves / 8;
  for (int i = tid; i < n; i += nthreads) d[i] = s[i];
}

// ---------------------------------------------------------------------------
// Utility kernels
// ---------------------------------------------------------------------------
__global__ void zero_k(float* __restrict__ p, int n) {
  int i = blockIdx.x * blockDim.x + threadIdx.x;
  if (i < n) p[i] = 0.0f;
}

__global__ void cvt_f16_k(const float* __restrict__ in, _Float16* __restrict__ out, int n) {
  int i = blockIdx.x * blockDim.x + threadIdx.x;
  if (i < n) out[i] = (_Float16)in[i];
}

// Wm1 is (64 x 129); keep K=0..127 as f16 (column 128, the t-feature column,
// is applied in f32 as a rank-1 update).
__global__ void cvt_wm1_k(const float* __restrict__ in, _Float16* __restrict__ out) {
  int i = blockIdx.x * blockDim.x + threadIdx.x;
  if (i < HIDDIM * 128) {
    int n = i >> 7, k = i & 127;
    out[i] = (_Float16)in[n * 129 + k];
  }
}

__global__ void deg_k(const int* __restrict__ dst, float* __restrict__ cnt, int ne) {
  int e = blockIdx.x * blockDim.x + threadIdx.x;
  if (e < ne) atomicAdd(&cnt[dst[e]], 1.0f);
}

// Scatter-add rows of h (D floats) into agg[dst]. One thread = one float4 chunk.
template <int D>
__global__ void scatter_k(const int* __restrict__ src, const int* __restrict__ dst,
                          const float* __restrict__ h, float* __restrict__ agg, int ne) {
  constexpr int C = D / 4;
  long long idx = (long long)blockIdx.x * blockDim.x + threadIdx.x;
  if (idx >= (long long)ne * C) return;
  int e = (int)(idx / C);
  int c = (int)(idx % C);
  int s = src[e], d = dst[e];
  const float4 val = reinterpret_cast<const float4*>(h + (size_t)s * D)[c];
  float* o = agg + (size_t)d * D + (size_t)c * 4;
  atomicAdd(o + 0, val.x);
  atomicAdd(o + 1, val.y);
  atomicAdd(o + 2, val.z);
  atomicAdd(o + 3, val.w);
}

// ---------------------------------------------------------------------------
// Layer 1: h1 = relu( (agg/cnt) @ W1l^T + b1 + x @ W1r^T )   K=128, N=64
// One wave per 16-node tile; weights staged in LDS per block.
// ---------------------------------------------------------------------------
__global__ void sage1_wmma_k(const float* __restrict__ agg, const float* __restrict__ cnt,
                             const float* __restrict__ x,
                             const _Float16* __restrict__ w1l, const _Float16* __restrict__ w1r,
                             const float* __restrict__ b1, float* __restrict__ h1) {
  __shared__ _Float16 swl[HIDDIM * INDIM];   // 16 KB
  __shared__ _Float16 swr[HIDDIM * INDIM];   // 16 KB
  stage_lds(swl, w1l, HIDDIM * INDIM, threadIdx.x, blockDim.x);
  stage_lds(swr, w1r, HIDDIM * INDIM, threadIdx.x, blockDim.x);
  __syncthreads();

  int wid  = (int)((blockIdx.x * blockDim.x + threadIdx.x) >> 5);
  int lane = threadIdx.x & 31;
  int m0 = wid * 16;
  if (m0 >= NNODES) return;               // wave-uniform (after barrier)
  int sub = lane & 15, half = lane >> 4;

  int row = m0 + sub;                      // NNODES % 16 == 0 -> always valid
  float inv = 1.0f / fmaxf(cnt[row], 1.0f);
  const float* arow = agg + (size_t)row * INDIM;
  const float* xrow = x   + (size_t)row * INDIM;

  v8f acc[4] = {};
#pragma unroll
  for (int kb = 0; kb < INDIM; kb += 32) {
    v16h a1 = frag_f32(arow, kb, half, inv);
    v16h a2 = frag_f32(xrow, kb, half, 1.0f);
#pragma unroll
    for (int nt = 0; nt < 4; ++nt) {
      v16h bl = frag_f16(swl + (size_t)(nt * 16 + sub) * INDIM, kb, half);
      v16h br = frag_f16(swr + (size_t)(nt * 16 + sub) * INDIM, kb, half);
      acc[nt] = __builtin_amdgcn_wmma_f32_16x16x32_f16(false, a1, false, bl,
                                                       (short)0, acc[nt], false, false);
      acc[nt] = __builtin_amdgcn_wmma_f32_16x16x32_f16(false, a2, false, br,
                                                       (short)0, acc[nt], false, false);
    }
  }
  // D layout: acc[nt][v] -> out[m0 + v + 8*half][nt*16 + sub]
#pragma unroll
  for (int nt = 0; nt < 4; ++nt) {
    int n = nt * 16 + sub;
    float bias = b1[n];
#pragma unroll
    for (int v = 0; v < 8; ++v) {
      int m = m0 + v + 8 * half;
      h1[(size_t)m * HIDDIM + n] = fmaxf(acc[nt][v] + bias, 0.0f);
    }
  }
}

// ---------------------------------------------------------------------------
// Layer 2: h2 = relu( (agg2/cnt) @ W2l^T + b2 + h1 @ W2r^T ), stored fp16.
// ---------------------------------------------------------------------------
__global__ void sage2_wmma_k(const float* __restrict__ agg, const float* __restrict__ cnt,
                             const float* __restrict__ h1,
                             const _Float16* __restrict__ w2l, const _Float16* __restrict__ w2r,
                             const float* __restrict__ b2, _Float16* __restrict__ h2) {
  __shared__ _Float16 swl[HIDDIM * HIDDIM];  // 8 KB
  __shared__ _Float16 swr[HIDDIM * HIDDIM];  // 8 KB
  stage_lds(swl, w2l, HIDDIM * HIDDIM, threadIdx.x, blockDim.x);
  stage_lds(swr, w2r, HIDDIM * HIDDIM, threadIdx.x, blockDim.x);
  __syncthreads();

  int wid  = (int)((blockIdx.x * blockDim.x + threadIdx.x) >> 5);
  int lane = threadIdx.x & 31;
  int m0 = wid * 16;
  if (m0 >= NNODES) return;
  int sub = lane & 15, half = lane >> 4;

  int row = m0 + sub;
  float inv = 1.0f / fmaxf(cnt[row], 1.0f);
  const float* arow = agg + (size_t)row * HIDDIM;
  const float* hrow = h1  + (size_t)row * HIDDIM;

  v8f acc[4] = {};
#pragma unroll
  for (int kb = 0; kb < HIDDIM; kb += 32) {
    v16h a1 = frag_f32(arow, kb, half, inv);
    v16h a2 = frag_f32(hrow, kb, half, 1.0f);
#pragma unroll
    for (int nt = 0; nt < 4; ++nt) {
      v16h bl = frag_f16(swl + (size_t)(nt * 16 + sub) * HIDDIM, kb, half);
      v16h br = frag_f16(swr + (size_t)(nt * 16 + sub) * HIDDIM, kb, half);
      acc[nt] = __builtin_amdgcn_wmma_f32_16x16x32_f16(false, a1, false, bl,
                                                       (short)0, acc[nt], false, false);
      acc[nt] = __builtin_amdgcn_wmma_f32_16x16x32_f16(false, a2, false, br,
                                                       (short)0, acc[nt], false, false);
    }
  }
#pragma unroll
  for (int nt = 0; nt < 4; ++nt) {
    int n = nt * 16 + sub;
    float bias = b2[n];
#pragma unroll
    for (int v = 0; v < 8; ++v) {
      int m = m0 + v + 8 * half;
      h2[(size_t)m * HIDDIM + n] = (_Float16)fmaxf(acc[nt][v] + bias, 0.0f);
    }
  }
}

// ---------------------------------------------------------------------------
// MLP head: z = [h2[u] | h2[v] | t] (129) -> relu(z@Wm1^T + bm1) @ Wm2^T + bm2
// One wave per 16-query tile; Wm1 (f16, K=128) staged in LDS; t-column in f32.
// Final 64-wide dot with Wm2 reduced via shfl_xor within half-waves.
// ---------------------------------------------------------------------------
__global__ void mlp_wmma_k(const _Float16* __restrict__ h2, const int* __restrict__ uv,
                           const float* __restrict__ tfeat,
                           const _Float16* __restrict__ wm1h, const float* __restrict__ wm1f,
                           const float* __restrict__ bm1, const float* __restrict__ wm2,
                           const float* __restrict__ bm2, float* __restrict__ y) {
  __shared__ _Float16 swm[HIDDIM * 128];     // 16 KB
  stage_lds(swm, wm1h, HIDDIM * 128, threadIdx.x, blockDim.x);
  __syncthreads();

  int wid  = (int)((blockIdx.x * blockDim.x + threadIdx.x) >> 5);
  int lane = threadIdx.x & 31;
  int q0 = wid * 16;
  if (q0 >= NQUERY) return;
  int sub = lane & 15, half = lane >> 4;

  int q = q0 + sub;                        // NQUERY % 16 == 0
  int un = uv[2 * q + 0];
  int vn = uv[2 * q + 1];
  const _Float16* pu = h2 + (size_t)un * HIDDIM;
  const _Float16* pv = h2 + (size_t)vn * HIDDIM;

  v8f acc[4] = {};
#pragma unroll
  for (int kb = 0; kb < HIDDIM; kb += 32) {   // h2[u] part: z-K = kb
    v16h a = frag_f16(pu, kb, half);
#pragma unroll
    for (int nt = 0; nt < 4; ++nt) {
      v16h b = frag_f16(swm + (size_t)(nt * 16 + sub) * 128, kb, half);
      acc[nt] = __builtin_amdgcn_wmma_f32_16x16x32_f16(false, a, false, b,
                                                       (short)0, acc[nt], false, false);
    }
  }
#pragma unroll
  for (int kb = 0; kb < HIDDIM; kb += 32) {   // h2[v] part: z-K = 64 + kb
    v16h a = frag_f16(pv, kb, half);
#pragma unroll
    for (int nt = 0; nt < 4; ++nt) {
      v16h b = frag_f16(swm + (size_t)(nt * 16 + sub) * 128, 64 + kb, half);
      acc[nt] = __builtin_amdgcn_wmma_f32_16x16x32_f16(false, a, false, b,
                                                       (short)0, acc[nt], false, false);
    }
  }

  float tq[8];
#pragma unroll
  for (int v = 0; v < 8; ++v) tq[v] = tfeat[q0 + v + 8 * half];

  float p[8];
#pragma unroll
  for (int v = 0; v < 8; ++v) p[v] = 0.0f;
#pragma unroll
  for (int nt = 0; nt < 4; ++nt) {
    int n = nt * 16 + sub;
    float wlast = wm1f[n * 129 + 128];       // t-feature column
    float bias  = bm1[n];
    float w2    = wm2[n];
#pragma unroll
    for (int v = 0; v < 8; ++v) {
      float s = acc[nt][v] + tq[v] * wlast + bias;
      p[v] += fmaxf(s, 0.0f) * w2;
    }
  }
  // Reduce over n (16 lanes of each half-wave hold distinct n)
#pragma unroll
  for (int v = 0; v < 8; ++v) {
#pragma unroll
    for (int mask = 1; mask < 16; mask <<= 1)
      p[v] += __shfl_xor(p[v], mask, 32);
  }
  if (sub == 0) {
    float b = bm2[0];
#pragma unroll
    for (int v = 0; v < 8; ++v) y[q0 + v + 8 * half] = p[v] + b;
  }
}

// ---------------------------------------------------------------------------
// Host launcher
// ---------------------------------------------------------------------------
extern "C" void kernel_launch(void* const* d_in, const int* in_sizes, int n_in,
                              void* d_out, int out_size, void* d_ws, size_t ws_size,
                              hipStream_t stream) {
  const float* x    = (const float*)d_in[0];
  const int*   ei   = (const int*)  d_in[1];   // (2, NEDGES): src row then dst row
  const int*   uv   = (const int*)  d_in[2];   // (NQUERY, 2)
  const float* tft  = (const float*)d_in[3];
  const float* W1l  = (const float*)d_in[4];
  const float* b1   = (const float*)d_in[5];
  const float* W1r  = (const float*)d_in[6];
  const float* W2l  = (const float*)d_in[7];
  const float* b2   = (const float*)d_in[8];
  const float* W2r  = (const float*)d_in[9];
  const float* Wm1  = (const float*)d_in[10];
  const float* bm1  = (const float*)d_in[11];
  const float* Wm2  = (const float*)d_in[12];
  const float* bm2  = (const float*)d_in[13];
  float* y = (float*)d_out;

  const int* src = ei;
  const int* dst = ei + NEDGES;

  // Workspace carve-up (256B aligned)
  char* p = (char*)d_ws;
  size_t off = 0;
  auto carve = [&](size_t bytes) -> char* {
    char* r = p + off;
    off += (bytes + 255) & ~(size_t)255;
    return r;
  };
  float*    cnt   = (float*)carve(sizeof(float) * NNODES);
  float*    agg1  = (float*)carve(sizeof(float) * (size_t)NNODES * INDIM);
  float*    h1    = (float*)carve(sizeof(float) * (size_t)NNODES * HIDDIM);
  float*    agg2  = (float*)carve(sizeof(float) * (size_t)NNODES * HIDDIM);
  _Float16* h2    = (_Float16*)carve(sizeof(_Float16) * (size_t)NNODES * HIDDIM);
  _Float16* w1lh  = (_Float16*)carve(sizeof(_Float16) * HIDDIM * INDIM);
  _Float16* w1rh  = (_Float16*)carve(sizeof(_Float16) * HIDDIM * INDIM);
  _Float16* w2lh  = (_Float16*)carve(sizeof(_Float16) * HIDDIM * HIDDIM);
  _Float16* w2rh  = (_Float16*)carve(sizeof(_Float16) * HIDDIM * HIDDIM);
  _Float16* wm1h  = (_Float16*)carve(sizeof(_Float16) * HIDDIM * 128);
  (void)ws_size; (void)n_in; (void)in_sizes; (void)out_size;

  const int T = 256;
  auto blocks = [](long long n, int t) { return (unsigned)((n + t - 1) / t); };

  // 1) zero accumulation buffers (every launch: graph replay reuses ws)
  zero_k<<<blocks(NNODES, T), T, 0, stream>>>(cnt, NNODES);
  zero_k<<<blocks((long long)NNODES * INDIM, T), T, 0, stream>>>(agg1, NNODES * INDIM);
  zero_k<<<blocks((long long)NNODES * HIDDIM, T), T, 0, stream>>>(agg2, NNODES * HIDDIM);

  // 2) weight conversion to f16
  cvt_f16_k<<<blocks(HIDDIM * INDIM, T), T, 0, stream>>>(W1l, w1lh, HIDDIM * INDIM);
  cvt_f16_k<<<blocks(HIDDIM * INDIM, T), T, 0, stream>>>(W1r, w1rh, HIDDIM * INDIM);
  cvt_f16_k<<<blocks(HIDDIM * HIDDIM, T), T, 0, stream>>>(W2l, w2lh, HIDDIM * HIDDIM);
  cvt_f16_k<<<blocks(HIDDIM * HIDDIM, T), T, 0, stream>>>(W2r, w2rh, HIDDIM * HIDDIM);
  cvt_wm1_k<<<blocks(HIDDIM * 128, T), T, 0, stream>>>(Wm1, wm1h);

  // 3) degrees
  deg_k<<<blocks(NEDGES, T), T, 0, stream>>>(dst, cnt, NEDGES);

  // 4) layer-1 scatter + WMMA GEMM
  scatter_k<INDIM><<<blocks((long long)NEDGES * (INDIM / 4), T), T, 0, stream>>>(
      src, dst, x, agg1, NEDGES);
  {
    int tiles = NNODES / 16;                       // 3125 waves
    sage1_wmma_k<<<blocks((long long)tiles * 32, T), T, 0, stream>>>(
        agg1, cnt, x, w1lh, w1rh, b1, h1);
  }

  // 5) layer-2 scatter + WMMA GEMM (h2 stored fp16 for cheap gathers)
  scatter_k<HIDDIM><<<blocks((long long)NEDGES * (HIDDIM / 4), T), T, 0, stream>>>(
      src, dst, h1, agg2, NEDGES);
  {
    int tiles = NNODES / 16;
    sage2_wmma_k<<<blocks((long long)tiles * 32, T), T, 0, stream>>>(
        agg2, cnt, h1, w2lh, w2rh, b2, h2);
  }

  // 6) MLP head over 400K query pairs (WMMA, K=128 f16 + f32 t-column)
  {
    int tiles = NQUERY / 16;                       // 25000 waves
    mlp_wmma_k<<<blocks((long long)tiles * 32, T), T, 0, stream>>>(
        h2, uv, tft, wm1h, Wm1, bm1, Wm2, bm2, y);
  }
}
